// AdaptiveJacobianPrunedViT_18846316494929
// MI455X (gfx1250) — matrix operand order repaired and checked by reading, not compiled, not run on hardware
//
#include <hip/hip_runtime.h>
#include <math.h>

// ---------------------------------------------------------------------------
// Types for CDNA5 WMMA (wave32): v_wmma_f32_16x16x32_bf16
// ---------------------------------------------------------------------------
typedef __bf16 v16bf __attribute__((ext_vector_type(16)));
typedef __bf16 v2bf  __attribute__((ext_vector_type(2)));
typedef float  v8f   __attribute__((ext_vector_type(8)));
typedef unsigned int uint4v __attribute__((ext_vector_type(4)));

#define L_   12
#define D_   768
#define H_   12
#define HD_  64
#define MLP_ 3072
#define NC_  1000
#define P_   16
#define IMG_ 224
#define B_   16
#define GRID_ 14
#define N0_  196
#define S0_  197

// Native bf16 conversion: let the backend pick the hardware cvt form.
static __device__ __forceinline__ unsigned int pk2(float a, float b) {
  union { v2bf v; unsigned int u; } c;
  c.v[0] = (__bf16)a;
  c.v[1] = (__bf16)b;
  return c.u;
}

union FragU {
  unsigned int   u[8];
  unsigned short h[16];
  __bf16         bh[16];
  v16bf          v;
};

static __device__ __forceinline__ void pack4(FragU& f, int o, float4 v) {
  f.bh[o * 2]     = (__bf16)v.x;
  f.bh[o * 2 + 1] = (__bf16)v.y;
  f.bh[o * 2 + 2] = (__bf16)v.z;
  f.bh[o * 2 + 3] = (__bf16)v.w;
}
static __device__ __forceinline__ void z4(bool keep, float4& v) {
  v.x = keep ? v.x : 0.0f; v.y = keep ? v.y : 0.0f;
  v.z = keep ? v.z : 0.0f; v.w = keep ? v.w : 0.0f;
}

// ---------------------------------------------------------------------------
// Tiled bf16 WMMA GEMM: C[M,N] = epilogue(A[M,K] @ W[N,K]^T + bias [+resid])
// Block 256 thr = 8 waves; tile 128x128x32; wave tile 32x64 (2x4 WMMA tiles).
// Double-buffered LDS; register-staged b128 global loads (row-clamped,
// branch-free); fp32 -> bf16 native casts on the LDS store; fragment reads
// are aligned b128 DS loads. Requires K % 32 == 0 (true for all launches).
// act: 0 = none, 1 = exact GELU
// ---------------------------------------------------------------------------
#define BM 128
#define BN 128
#define BK 32
#define LDK 40   // padded LDS row stride (ushorts); 80B rows keep b128 frags aligned

static __device__ __forceinline__ void load_tile(
    const float* __restrict__ A, const float* __restrict__ W,
    int M, int N, int K, int m0, int n0, int k0, int tid,
    float4 ra[4], float4 rw[4])
{
#pragma unroll
  for (int e = 0; e < 4; ++e) {
    int flat4 = tid + e * 256;          // 0..1023 float4 slots
    int r = flat4 >> 3, c4 = flat4 & 7; // row 0..127, 4-float col group 0..7
    int gm = m0 + r;  bool okA = gm < M;
    int gn = n0 + r;  bool okW = gn < N;
    ra[e] = *(const float4*)(A + (size_t)(okA ? gm : 0) * K + k0 + c4 * 4);
    rw[e] = *(const float4*)(W + (size_t)(okW ? gn : 0) * K + k0 + c4 * 4);
    z4(okA, ra[e]);
    z4(okW, rw[e]);
  }
}

static __device__ __forceinline__ void store_tile(
    unsigned short* __restrict__ Asb, unsigned short* __restrict__ Wsb,
    int tid, const float4 ra[4], const float4 rw[4])
{
#pragma unroll
  for (int e = 0; e < 4; ++e) {
    int flat4 = tid + e * 256;
    int r = flat4 >> 3, c4 = flat4 & 7;
    uint2 pa, pw;
    pa.x = pk2(ra[e].x, ra[e].y); pa.y = pk2(ra[e].z, ra[e].w);
    pw.x = pk2(rw[e].x, rw[e].y); pw.y = pk2(rw[e].z, rw[e].w);
    *(uint2*)&Asb[r * LDK + c4 * 4] = pa;
    *(uint2*)&Wsb[r * LDK + c4 * 4] = pw;
  }
}

__global__ __launch_bounds__(256)
void gemm_bf16(const float* __restrict__ A, const float* __restrict__ W,
               const float* __restrict__ bias, const float* __restrict__ resid,
               float* __restrict__ C, int M, int N, int K, int act)
{
  __shared__ unsigned short As[2][BM * LDK];
  __shared__ unsigned short Ws[2][BN * LDK];
  const int tid  = threadIdx.x;
  const int lane = tid & 31;
  const int wave = tid >> 5;
  const int wm   = wave >> 1;   // 0..3
  const int wn   = wave & 1;    // 0..1
  const int m0   = blockIdx.y * BM;
  const int n0   = blockIdx.x * BN;
  const int half = lane >> 4;
  const int l16  = lane & 15;

  v8f acc[2][4];
#pragma unroll
  for (int i = 0; i < 2; ++i)
#pragma unroll
    for (int j = 0; j < 4; ++j)
#pragma unroll
      for (int r = 0; r < 8; ++r) acc[i][j][r] = 0.0f;

  float4 ra[4], rw[4];
  load_tile(A, W, M, N, K, m0, n0, 0, tid, ra, rw);
  store_tile(&As[0][0], &Ws[0][0], tid, ra, rw);
  __syncthreads();

  int cur = 0;
  for (int k0 = 0; k0 < K; k0 += BK) {
    const int kn = k0 + BK;
    const bool hasNext = kn < K;
    if (hasNext) {
      load_tile(A, W, M, N, K, m0, n0, kn, tid, ra, rw);
      if (kn + BK < K) {  // prefetch two tiles ahead (global_prefetch_b8)
        int r = tid >> 3;
        __builtin_prefetch(A + (size_t)((m0 + r) < M ? (m0 + r) : 0) * K + kn + BK, 0, 1);
        __builtin_prefetch(W + (size_t)((n0 + r) < N ? (n0 + r) : 0) * K + kn + BK, 0, 1);
      }
    }

    // ---- fragments from current LDS buffer (b128 DS loads) ----
    const unsigned short* Ac = &As[cur][0];
    const unsigned short* Wc = &Ws[cur][0];
    FragU a[2], b[4];
#pragma unroll
    for (int mt = 0; mt < 2; ++mt) {
      int mrow = wm * 32 + mt * 16 + l16;
      *(uint4v*)&a[mt].u[0] = *(const uint4v*)&Ac[mrow * LDK + half * 8];
      *(uint4v*)&a[mt].u[4] = *(const uint4v*)&Ac[mrow * LDK + 16 + half * 8];
    }
#pragma unroll
    for (int nt = 0; nt < 4; ++nt) {
      int ncol = wn * 64 + nt * 16 + l16;
      *(uint4v*)&b[nt].u[0] = *(const uint4v*)&Wc[ncol * LDK + half * 8];
      *(uint4v*)&b[nt].u[4] = *(const uint4v*)&Wc[ncol * LDK + 16 + half * 8];
    }

#pragma unroll
    for (int mt = 0; mt < 2; ++mt)
#pragma unroll
      for (int nt = 0; nt < 4; ++nt)
        acc[mt][nt] = __builtin_amdgcn_wmma_f32_16x16x32_bf16(
            false, a[mt].v, false, b[nt].v, (short)0, acc[mt][nt], false, false);

    if (hasNext) {
      store_tile(&As[cur ^ 1][0], &Ws[cur ^ 1][0], tid, ra, rw);
      __syncthreads();
      cur ^= 1;
    }
  }

  // ---- epilogue: bias / GELU / residual ----
#pragma unroll
  for (int mt = 0; mt < 2; ++mt)
#pragma unroll
    for (int nt = 0; nt < 4; ++nt)
#pragma unroll
      for (int r = 0; r < 8; ++r) {
        int row = m0 + wm * 32 + mt * 16 + half * 8 + r;
        int col = n0 + wn * 64 + nt * 16 + l16;
        if (row < M && col < N) {
          float v = acc[mt][nt][r];
          if (bias)  v += bias[col];
          if (act == 1) v = 0.5f * v * (1.0f + erff(v * 0.7071067811865475f));
          if (resid) v += resid[(size_t)row * N + col];
          C[(size_t)row * N + col] = v;
        }
      }
}

// ---------------------------------------------------------------------------
// Attention QK^T (one wave per 16x16 score tile), scores * HD^-0.5
// qkv layout: [B*S, 2304] = q(+0) | k(+768) | v(+1536), each [H=12][HD=64]
// Writes padded logits over [Spad x Spad] (Spad = S rounded up to 32).
// ---------------------------------------------------------------------------
__global__ __launch_bounds__(32)
void attn_qk(const float* __restrict__ qkv, float* __restrict__ Ab, int S, int Spad)
{
  const int qt = blockIdx.x, kt = blockIdx.y, bh = blockIdx.z;
  const int b = bh / H_, h = bh % H_;
  const int lane = threadIdx.x, half = lane >> 4, l16 = lane & 15;
  const int q0 = qt * 16, k0 = kt * 16;

  v8f acc;
#pragma unroll
  for (int i = 0; i < 8; ++i) acc[i] = 0.0f;

  const int m = q0 + l16, n = k0 + l16;
  const bool okm = m < S, okn = n < S;
  const float4* q4 = (const float4*)(qkv + (size_t)(b * S + (okm ? m : 0)) * 2304 + h * 64);
  const float4* k4 = (const float4*)(qkv + (size_t)(b * S + (okn ? n : 0)) * 2304 + 768 + h * 64);

#pragma unroll
  for (int dB = 0; dB < 2; ++dB) {  // two 32-wide K chunks of HD=64
    FragU a, bf;
    float4 a0 = q4[dB * 8 + half * 2];
    float4 a1 = q4[dB * 8 + half * 2 + 1];
    float4 a2 = q4[dB * 8 + 4 + half * 2];
    float4 a3 = q4[dB * 8 + 5 + half * 2];
    z4(okm, a0); z4(okm, a1); z4(okm, a2); z4(okm, a3);
    pack4(a, 0, a0); pack4(a, 2, a1); pack4(a, 4, a2); pack4(a, 6, a3);

    float4 b0 = k4[dB * 8 + half * 2];
    float4 b1 = k4[dB * 8 + half * 2 + 1];
    float4 b2 = k4[dB * 8 + 4 + half * 2];
    float4 b3 = k4[dB * 8 + 5 + half * 2];
    z4(okn, b0); z4(okn, b1); z4(okn, b2); z4(okn, b3);
    pack4(bf, 0, b0); pack4(bf, 2, b1); pack4(bf, 4, b2); pack4(bf, 6, b3);

    acc = __builtin_amdgcn_wmma_f32_16x16x32_bf16(false, a.v, false, bf.v,
                                                  (short)0, acc, false, false);
  }

  const float scale = 0.125f;  // 64^-0.5
#pragma unroll
  for (int r = 0; r < 8; ++r) {
    int q = q0 + half * 8 + r;
    int k = k0 + l16;
    Ab[((size_t)bh * Spad + q) * Spad + k] = acc[r] * scale;
  }
}

// ---------------------------------------------------------------------------
// Row softmax over k < S; zeroes pad columns [S, Spad) so AV can run
// unconditional full-width K tiles. One wave per row (wave32 shuffles).
// ---------------------------------------------------------------------------
__global__ __launch_bounds__(256)
void softmax_rows(float* __restrict__ Ab, int S, int Spad, int nRows)
{
  int rowIdx = blockIdx.x * 8 + (int)(threadIdx.x >> 5);
  if (rowIdx >= nRows) return;
  int lane = threadIdx.x & 31;
  int bh = rowIdx / S, q = rowIdx % S;
  float* row = Ab + ((size_t)bh * Spad + q) * Spad;

  float mx = -3.0e38f;
  for (int k = lane; k < S; k += 32) mx = fmaxf(mx, row[k]);
#pragma unroll
  for (int o = 16; o >= 1; o >>= 1) mx = fmaxf(mx, __shfl_xor(mx, o, 32));
  float sum = 0.0f;
  for (int k = lane; k < S; k += 32) { float e = __expf(row[k] - mx); row[k] = e; sum += e; }
#pragma unroll
  for (int o = 16; o >= 1; o >>= 1) sum += __shfl_xor(sum, o, 32);
  float inv = 1.0f / sum;
  for (int k = lane; k < S; k += 32) row[k] *= inv;
  for (int k = S + lane; k < Spad; k += 32) row[k] = 0.0f;  // zero pads
}

// ---------------------------------------------------------------------------
// O = softmax(A) @ V  (one wave per 16(q) x 16(d) tile, WMMA over padded K)
// P fragments: vectorized b128 loads (pads are zero); V: row-guarded gather
// with native bf16 casts. O heads-merged: [B*S, 768]
// ---------------------------------------------------------------------------
__global__ __launch_bounds__(32)
void attn_av(const float* __restrict__ Ab, const float* __restrict__ qkv,
             float* __restrict__ O, int S, int Spad)
{
  const int qt = blockIdx.x, dt = blockIdx.y, bh = blockIdx.z;
  const int b = bh / H_, h = bh % H_;
  const int lane = threadIdx.x, half = lane >> 4, l16 = lane & 15;
  const int q0 = qt * 16, d0 = dt * 16;

  v8f acc;
#pragma unroll
  for (int i = 0; i < 8; ++i) acc[i] = 0.0f;

  const int m = q0 + l16;
  const int n = d0 + l16;
  // rows m >= S only influence output rows >= S, which are never stored
  const float4* p4 = (const float4*)(Ab + ((size_t)bh * Spad + (m < S ? m : 0)) * Spad);

  for (int k0 = 0; k0 < Spad; k0 += 32) {
    FragU a, bf;
    const int f0 = k0 >> 2;
    float4 a0 = p4[f0 + half * 2];
    float4 a1 = p4[f0 + half * 2 + 1];
    float4 a2 = p4[f0 + 4 + half * 2];
    float4 a3 = p4[f0 + 5 + half * 2];
    pack4(a, 0, a0); pack4(a, 2, a1); pack4(a, 4, a2); pack4(a, 6, a3);

#pragma unroll
    for (int i = 0; i < 16; ++i) {
      int kk = k0 + half * 8 + (i & 7) + ((i >> 3) << 4);
      bool ok = kk < S;
      float v = qkv[(size_t)(b * S + (ok ? kk : 0)) * 2304 + 1536 + h * 64 + n];
      bf.bh[i] = (__bf16)(ok ? v : 0.0f);
    }
    acc = __builtin_amdgcn_wmma_f32_16x16x32_bf16(false, a.v, false, bf.v,
                                                  (short)0, acc, false, false);
  }

#pragma unroll
  for (int r = 0; r < 8; ++r) {
    int q = q0 + half * 8 + r;
    if (q < S)
      O[(size_t)(b * S + q) * 768 + h * 64 + n] = acc[r];
  }
}

// ---------------------------------------------------------------------------
// LayerNorm over D=768, one block per row
// ---------------------------------------------------------------------------
__global__ __launch_bounds__(256)
void layernorm(const float* __restrict__ X, const float* __restrict__ g,
               const float* __restrict__ bta, float* __restrict__ Y)
{
  const int row = blockIdx.x;
  const int tid = threadIdx.x;
  const float* x = X + (size_t)row * 768;
  float s = 0.0f, ss = 0.0f;
  for (int i = tid; i < 768; i += 256) { float v = x[i]; s += v; ss += v * v; }
#pragma unroll
  for (int o = 16; o >= 1; o >>= 1) { s += __shfl_xor(s, o, 32); ss += __shfl_xor(ss, o, 32); }
  __shared__ float sh[16];
  int wave = tid >> 5, lane = tid & 31;
  if (lane == 0) { sh[wave] = s; sh[8 + wave] = ss; }
  __syncthreads();
  if (tid == 0) {
    float a = 0.0f, b2 = 0.0f;
    for (int w = 0; w < 8; ++w) { a += sh[w]; b2 += sh[8 + w]; }
    sh[0] = a; sh[8] = b2;
  }
  __syncthreads();
  float mu  = sh[0] * (1.0f / 768.0f);
  float var = sh[8] * (1.0f / 768.0f) - mu * mu;
  float inv = rsqrtf(var + 1e-6f);
  for (int i = tid; i < 768; i += 256)
    Y[(size_t)row * 768 + i] = (x[i] - mu) * inv * g[i] + bta[i];
}

// ---------------------------------------------------------------------------
// Patch extraction (im2col) matching OIHW conv, patch vec order (c,ph,pw)
// ---------------------------------------------------------------------------
__global__ void im2col(const float* __restrict__ x, float* __restrict__ pat)
{
  size_t idx = (size_t)blockIdx.x * blockDim.x + threadIdx.x;
  const size_t total = (size_t)B_ * N0_ * 768;
  if (idx >= total) return;
  int col = (int)(idx % 768);
  size_t row = idx / 768;
  int p = (int)(row % N0_); int b = (int)(row / N0_);
  int pw = col & 15, ph = (col >> 4) & 15, c = col >> 8;
  int py = p / GRID_, px = p % GRID_;
  pat[idx] = x[(((size_t)b * 3 + c) * IMG_ + py * 16 + ph) * IMG_ + px * 16 + pw];
}

__global__ void build_xt(const float* __restrict__ tok, const float* __restrict__ cls,
                         const float* __restrict__ pos, float* __restrict__ xt)
{
  int bs = blockIdx.x; int b = bs / S0_, s = bs % S0_;
  for (int d = threadIdx.x; d < 768; d += blockDim.x) {
    float v = (s == 0) ? cls[d] : tok[((size_t)b * N0_ + (s - 1)) * 768 + d];
    xt[((size_t)b * S0_ + s) * 768 + d] = v + pos[(size_t)s * 768 + d];
  }
}

// imp[j] = mean_{b,h}( (sum_q A[b,h,q,j]) * ||v[b,h,j]|| ), j = 1..S-1
__global__ __launch_bounds__(256)
void importance(const float* __restrict__ Ab, const float* __restrict__ qkv,
                float* __restrict__ imp, int S, int Spad)
{
  const int j = blockIdx.x + 1;
  const int tid = threadIdx.x;
  float accum = 0.0f;
  for (int bh = tid; bh < B_ * H_; bh += 256) {
    int b = bh / H_, h = bh % H_;
    float cs = 0.0f;
    const float* col = Ab + (size_t)bh * Spad * Spad + j;
    for (int q = 0; q < S; ++q) cs += col[(size_t)q * Spad];
    float vn = 0.0f;
    const float* vrow = qkv + (size_t)(b * S + j) * 2304 + 1536 + h * 64;
#pragma unroll 8
    for (int d = 0; d < 64; ++d) { float v = vrow[d]; vn += v * v; }
    accum += cs * sqrtf(vn);
  }
#pragma unroll
  for (int o = 16; o >= 1; o >>= 1) accum += __shfl_xor(accum, o, 32);
  __shared__ float sh[8];
  if ((tid & 31) == 0) sh[tid >> 5] = accum;
  __syncthreads();
  if (tid == 0) {
    float t = 0.0f;
    for (int w = 0; w < 8; ++w) t += sh[w];
    imp[j] = t * (1.0f / (B_ * H_));
  }
}

// top-tN selection + ascending-index keep list (keep[0] = CLS)
__global__ void select_topk(const float* __restrict__ imp, int* __restrict__ keep,
                            int Ntok, int tN)
{
  if (threadIdx.x != 0 || blockIdx.x != 0) return;
  bool used[224];
  for (int j = 0; j <= Ntok; ++j) used[j] = false;
  for (int t = 0; t < tN; ++t) {
    float best = -3.0e38f; int bi = -1;
    for (int j = 1; j <= Ntok; ++j)
      if (!used[j] && imp[j] > best) { best = imp[j]; bi = j; }
    if (bi >= 0) used[bi] = true;
  }
  int pos = 0; keep[pos++] = 0;
  for (int j = 1; j <= Ntok; ++j) if (used[j]) keep[pos++] = j;
}

__global__ void gather_tokens(const float* __restrict__ xin, const int* __restrict__ keep,
                              float* __restrict__ xout, int Sold, int Snew)
{
  int bi = blockIdx.x; int b = bi / Snew, i = bi % Snew;
  int src = keep[i];
  for (int d = threadIdx.x; d < 768; d += blockDim.x)
    xout[((size_t)b * Snew + i) * 768 + d] = xin[((size_t)b * Sold + src) * 768 + d];
}

__global__ void extract_cls(const float* __restrict__ xt, float* __restrict__ cls, int S)
{
  int b = blockIdx.x;
  for (int d = threadIdx.x; d < 768; d += blockDim.x)
    cls[(size_t)b * 768 + d] = xt[(size_t)(b * S) * 768 + d];
}

// ---------------------------------------------------------------------------
// Host orchestration
// ---------------------------------------------------------------------------
extern "C" void kernel_launch(void* const* d_in, const int* in_sizes, int n_in,
                              void* d_out, int out_size, void* d_ws, size_t ws_size,
                              hipStream_t stream)
{
  (void)in_sizes; (void)n_in; (void)out_size; (void)ws_size;
  const float* x       = (const float*)d_in[0];
  const float* patch_w = (const float*)d_in[1];
  const float* patch_b = (const float*)d_in[2];
  const float* cls_tok = (const float*)d_in[3];
  const float* pos_emb = (const float*)d_in[4];
  const float* ln1_g   = (const float*)d_in[5];
  const float* ln1_b   = (const float*)d_in[6];
  const float* qkv_w   = (const float*)d_in[7];
  const float* qkv_b   = (const float*)d_in[8];
  const float* proj_w  = (const float*)d_in[9];
  const float* proj_b  = (const float*)d_in[10];
  const float* ln2_g   = (const float*)d_in[11];
  const float* ln2_b   = (const float*)d_in[12];
  const float* fc1_w   = (const float*)d_in[13];
  const float* fc1_b   = (const float*)d_in[14];
  const float* fc2_w   = (const float*)d_in[15];
  const float* fc2_b   = (const float*)d_in[16];
  const float* norm_g  = (const float*)d_in[17];
  const float* norm_b  = (const float*)d_in[18];
  const float* head_w  = (const float*)d_in[19];
  const float* head_b  = (const float*)d_in[20];
  float* out = (float*)d_out;

  char* ws = (char*)d_ws;
  size_t off = 0;
  auto alloc = [&](size_t bytes) -> char* {
    char* p = ws + off;
    off = (off + bytes + 255) & ~(size_t)255;
    return p;
  };
  const size_t tokBytes = (size_t)B_ * S0_ * 768 * 4;
  float* xtA  = (float*)alloc(tokBytes);
  float* xtB  = (float*)alloc(tokBytes);
  float* xn   = (float*)alloc(tokBytes);
  float* qkv  = (float*)alloc((size_t)B_ * S0_ * 2304 * 4);
  float* Ab   = (float*)alloc((size_t)B_ * H_ * 224 * 224 * 4);  // stride Spad<=224
  float* Obuf = (float*)alloc(tokBytes);
  float* hbuf = (float*)alloc((size_t)B_ * S0_ * 3072 * 4);      // also im2col scratch
  float* imp  = (float*)alloc(256 * 4);
  int*   keep = (int*)alloc(256 * 4);
  float* clsb = (float*)alloc((size_t)B_ * 768 * 4);
  float* clsn = (float*)alloc((size_t)B_ * 768 * 4);

  // ---- patch embed: im2col -> GEMM -> assemble tokens (+cls, +pos) ----
  {
    size_t total = (size_t)B_ * N0_ * 768;
    im2col<<<(unsigned)((total + 255) / 256), 256, 0, stream>>>(x, hbuf);
    dim3 g((768 + BN - 1) / BN, (B_ * N0_ + BM - 1) / BM);
    gemm_bf16<<<g, 256, 0, stream>>>(hbuf, patch_w, patch_b, nullptr, xn,
                                     B_ * N0_, 768, 768, 0);
    build_xt<<<B_ * S0_, 256, 0, stream>>>(xn, cls_tok, pos_emb, xtA);
  }

  float* xt  = xtA;
  float* alt = xtB;
  int Ntok = N0_;

  // LN1 + QKV GEMM + QK^T + softmax for current xt; returns Spad
  auto attn_front = [&](int l, int S) -> int {
    int M = B_ * S;
    layernorm<<<M, 256, 0, stream>>>(xt, ln1_g + (size_t)l * 768,
                                     ln1_b + (size_t)l * 768, xn);
    dim3 g((2304 + BN - 1) / BN, (M + BM - 1) / BM);
    gemm_bf16<<<g, 256, 0, stream>>>(xn, qkv_w + (size_t)l * 2304 * 768,
                                     qkv_b + (size_t)l * 2304, nullptr, qkv,
                                     M, 2304, 768, 0);
    int Spad = (S + 31) & ~31;
    dim3 gq(Spad / 16, Spad / 16, B_ * H_);
    attn_qk<<<gq, 32, 0, stream>>>(qkv, Ab, S, Spad);
    int nRows = B_ * H_ * S;
    softmax_rows<<<(nRows + 7) / 8, 256, 0, stream>>>(Ab, S, Spad, nRows);
    return Spad;
  };

  for (int l = 0; l < L_; ++l) {
    // static pruning schedule (mirrors _target_tokens)
    double frac = (double)l / (double)(L_ - 1);
    double keepf = 1.0 - 0.6 * frac * frac;
    if (keepf < 0.0) keepf = 0.0;
    int tN = (int)(N0_ * keepf);
    if (tN < 16) tN = 16;

    if (Ntok > tN) {
      // redundant importance pass on pre-prune tokens (as in reference)
      int S = Ntok + 1;
      int Spad = attn_front(l, S);
      importance<<<S - 1, 256, 0, stream>>>(Ab, qkv, imp, S, Spad);
      select_topk<<<1, 1, 0, stream>>>(imp, keep, Ntok, tN);
      int Snew = tN + 1;
      gather_tokens<<<B_ * Snew, 256, 0, stream>>>(xt, keep, alt, S, Snew);
      float* t = xt; xt = alt; alt = t;
      Ntok = tN;
    }

    int S = Ntok + 1;
    int M = B_ * S;
    int Spad = attn_front(l, S);

    dim3 gav(Spad / 16, HD_ / 16, B_ * H_);
    attn_av<<<gav, 32, 0, stream>>>(Ab, qkv, Obuf, S, Spad);

    dim3 gp((768 + BN - 1) / BN, (M + BM - 1) / BM);
    gemm_bf16<<<gp, 256, 0, stream>>>(Obuf, proj_w + (size_t)l * 768 * 768,
                                      proj_b + (size_t)l * 768, xt, xt,
                                      M, 768, 768, 0);

    layernorm<<<M, 256, 0, stream>>>(xt, ln2_g + (size_t)l * 768,
                                     ln2_b + (size_t)l * 768, xn);
    dim3 g1((3072 + BN - 1) / BN, (M + BM - 1) / BM);
    gemm_bf16<<<g1, 256, 0, stream>>>(xn, fc1_w + (size_t)l * 3072 * 768,
                                      fc1_b + (size_t)l * 3072, nullptr, hbuf,
                                      M, 3072, 768, 1);
    dim3 g2((768 + BN - 1) / BN, (M + BM - 1) / BM);
    gemm_bf16<<<g2, 256, 0, stream>>>(hbuf, fc2_w + (size_t)l * 768 * 3072,
                                      fc2_b + (size_t)l * 768, xt, xt,
                                      M, 768, 3072, 0);
  }

  // ---- final LN on CLS rows + classification head ----
  int S = Ntok + 1;
  extract_cls<<<B_, 256, 0, stream>>>(xt, clsb, S);
  layernorm<<<B_, 256, 0, stream>>>(clsb, norm_g, norm_b, clsn);
  dim3 gh((NC_ + BN - 1) / BN, (B_ + BM - 1) / BM);
  gemm_bf16<<<gh, 256, 0, stream>>>(clsn, head_w, head_b, nullptr, out,
                                    B_, NC_, 768, 0);
}
